// MLPGraphNetwork_10247791968479
// MI455X (gfx1250) — matrix operand order repaired and checked by reading, not compiled, not run on hardware
//
#include <hip/hip_runtime.h>
#include <hip/hip_bf16.h>

typedef __attribute__((ext_vector_type(16))) __bf16 v16bf;
typedef __attribute__((ext_vector_type(8)))  float  v8f;

union BFrag { unsigned u[8]; v16bf v; };

__device__ __forceinline__ unsigned short f2bf(float f) {
  unsigned u = __float_as_uint(f);
  u += 0x7FFFu + ((u >> 16) & 1u);           // round-to-nearest-even
  return (unsigned short)(u >> 16);
}

__device__ __forceinline__ void pack4(unsigned short* dst, float4 v) {
  unsigned lo = (unsigned)f2bf(v.x) | ((unsigned)f2bf(v.y) << 16);
  unsigned hi = (unsigned)f2bf(v.z) | ((unsigned)f2bf(v.w) << 16);
  *(unsigned*)(dst)     = lo;   // 4-byte aligned (even element index, even strides)
  *(unsigned*)(dst + 2) = hi;
}

// A-fragment K offset for 16-bit A 16x32 (ISA 7.12.2)
__device__ __forceinline__ int klA(int v, int half) {
  return (v < 4 ? 2 * v : 16 + 2 * (v - 4)) + half * 8;
}

// ---------------------------------------------------------------------------
// Hidden layer: 64 rows (bf16 LDS) x [KD x 128] -> 64 x 128 bf16 LDS
// 8 waves; wave w owns col strip [16w,16w+16) across 4 row tiles.
// B fragment loaded once per k-step, reused by 4 WMMAs.
// ---------------------------------------------------------------------------
template<int KD>
__device__ __forceinline__ void layer64x128(
    const unsigned short* __restrict__ in, int sIn,
    unsigned short* __restrict__ out, int sOut,
    const unsigned* __restrict__ Wp, const float* __restrict__ bias,
    int w, int lane)
{
  const int half = lane >> 4;
  const int row  = lane & 15;
  const int cb   = w * 16;
  v8f acc0 = {}, acc1 = {}, acc2 = {}, acc3 = {};
#pragma unroll
  for (int k0 = 0; k0 < KD; k0 += 32) {
    BFrag B;
#pragma unroll
    for (int v = 0; v < 8; ++v)
      B.u[v] = Wp[((k0 + klA(v, half)) >> 1) * 128 + cb + row];
    BFrag A0, A1, A2, A3;
#pragma unroll
    for (int v = 0; v < 8; ++v) {
      const int ka = k0 + klA(v, half);
      A0.u[v] = *(const unsigned*)(in + (0 * 16 + row) * sIn + ka);
      A1.u[v] = *(const unsigned*)(in + (1 * 16 + row) * sIn + ka);
      A2.u[v] = *(const unsigned*)(in + (2 * 16 + row) * sIn + ka);
      A3.u[v] = *(const unsigned*)(in + (3 * 16 + row) * sIn + ka);
    }
    acc0 = __builtin_amdgcn_wmma_f32_16x16x32_bf16(false, A0.v, false, B.v, (short)0, acc0, false, false);
    acc1 = __builtin_amdgcn_wmma_f32_16x16x32_bf16(false, A1.v, false, B.v, (short)0, acc1, false, false);
    acc2 = __builtin_amdgcn_wmma_f32_16x16x32_bf16(false, A2.v, false, B.v, (short)0, acc2, false, false);
    acc3 = __builtin_amdgcn_wmma_f32_16x16x32_bf16(false, A3.v, false, B.v, (short)0, acc3, false, false);
  }
  const int n0 = cb + row;
  const float bb = bias[n0];
  v8f* accs[4] = { &acc0, &acc1, &acc2, &acc3 };
#pragma unroll
  for (int r = 0; r < 4; ++r) {
#pragma unroll
    for (int v = 0; v < 8; ++v) {
      const int m = r * 16 + half * 8 + v;
      float x = (*accs[r])[v] + bb;
      x = x > 0.f ? x : 0.f;
      out[m * sOut + n0] = f2bf(x);
    }
  }
}

// ---------------------------------------------------------------------------
// Edge MLP: 64 edges/block. in = [edge(32)|src(64)|dst(64)|g(64)] = 224
// 224->128->128->128->32 (final fp32, no relu) + agg scatter + col sums
// ---------------------------------------------------------------------------
__global__ __launch_bounds__(256) void edge_kernel(
    const float* __restrict__ nodes, const float* __restrict__ edges,
    const float* __restrict__ g,
    const int* __restrict__ senders, const int* __restrict__ receivers,
    const unsigned* __restrict__ Wp0, const unsigned* __restrict__ Wp1,
    const unsigned* __restrict__ Wp2, const unsigned* __restrict__ Wp3,
    const float* __restrict__ b0, const float* __restrict__ b1,
    const float* __restrict__ b2, const float* __restrict__ b3,
    float* __restrict__ out_edges, float* __restrict__ agg,
    float* __restrict__ edge_sum, int E)
{
  __shared__ __align__(16) unsigned short buf0[64 * 226];
  __shared__ __align__(16) unsigned short buf1[64 * 130];
  __shared__ __align__(16) unsigned short buf2[64 * 130];
  __shared__ float colsum[32];
  const int tid = threadIdx.x;
  const int e0  = blockIdx.x * 64;
  if (tid < 32) colsum[tid] = 0.f;

  // gather, float4 granularity: 224 cols = 56 quads (8 | 16 | 16 | 16)
  for (int idx = tid; idx < 64 * 56; idx += 256) {
    const int r = idx / 56, q = idx % 56;
    const int e  = e0 + r;
    const int ec = e < E ? e : 0;
    float4 v;
    if (q < 8)       v = ((const float4*)(edges + (size_t)ec * 32))[q];
    else if (q < 24) v = ((const float4*)(nodes + (size_t)senders[ec] * 64))[q - 8];
    else if (q < 40) v = ((const float4*)(nodes + (size_t)receivers[ec] * 64))[q - 24];
    else             v = ((const float4*)g)[q - 40];
    pack4(&buf0[r * 226 + q * 4], v);
  }
  __syncthreads();

  const int w = tid >> 5, lane = tid & 31;
  layer64x128<224>(buf0, 226, buf1, 130, Wp0, b0, w, lane); __syncthreads();
  layer64x128<128>(buf1, 130, buf2, 130, Wp1, b1, w, lane); __syncthreads();
  layer64x128<128>(buf2, 130, buf1, 130, Wp2, b2, w, lane); __syncthreads();

  // final layer: 128 -> 32, fp32 output, no relu. Waves 0,1 active (cb<32).
  {
    const int half = lane >> 4, row = lane & 15;
    const int cb = w * 16;
    if (cb < 32) {
      v8f acc0 = {}, acc1 = {}, acc2 = {}, acc3 = {};
#pragma unroll
      for (int k0 = 0; k0 < 128; k0 += 32) {
        BFrag B;
#pragma unroll
        for (int v = 0; v < 8; ++v)
          B.u[v] = Wp3[((k0 + klA(v, half)) >> 1) * 32 + cb + row];
        BFrag A0, A1, A2, A3;
#pragma unroll
        for (int v = 0; v < 8; ++v) {
          const int ka = k0 + klA(v, half);
          A0.u[v] = *(const unsigned*)(buf1 + (0 * 16 + row) * 130 + ka);
          A1.u[v] = *(const unsigned*)(buf1 + (1 * 16 + row) * 130 + ka);
          A2.u[v] = *(const unsigned*)(buf1 + (2 * 16 + row) * 130 + ka);
          A3.u[v] = *(const unsigned*)(buf1 + (3 * 16 + row) * 130 + ka);
        }
        acc0 = __builtin_amdgcn_wmma_f32_16x16x32_bf16(false, A0.v, false, B.v, (short)0, acc0, false, false);
        acc1 = __builtin_amdgcn_wmma_f32_16x16x32_bf16(false, A1.v, false, B.v, (short)0, acc1, false, false);
        acc2 = __builtin_amdgcn_wmma_f32_16x16x32_bf16(false, A2.v, false, B.v, (short)0, acc2, false, false);
        acc3 = __builtin_amdgcn_wmma_f32_16x16x32_bf16(false, A3.v, false, B.v, (short)0, acc3, false, false);
      }
      const int n0 = cb + row;
      const float bb = b3[n0];
      float s = 0.f;
      v8f* accs[4] = { &acc0, &acc1, &acc2, &acc3 };
#pragma unroll
      for (int r = 0; r < 4; ++r) {
#pragma unroll
        for (int v = 0; v < 8; ++v) {
          const int m = r * 16 + half * 8 + v;
          const int e = e0 + m;
          if (e < E) {
            const float x = (*accs[r])[v] + bb;
            out_edges[(size_t)e * 32 + n0] = x;
            atomicAdd(&agg[(size_t)receivers[e] * 32 + n0], x);
            s += x;
          }
        }
      }
      atomicAdd(&colsum[n0], s);
    }
  }
  __syncthreads();
  if (tid < 32) atomicAdd(&edge_sum[tid], colsum[tid]);
}

// ---------------------------------------------------------------------------
// Node MLP: 64 nodes/block. in = [node(64)|agg(32)|g(64)] = 160
// 160->128->128->128->64 (final fp32, no relu) + col sums
// ---------------------------------------------------------------------------
__global__ __launch_bounds__(256) void node_kernel(
    const float* __restrict__ nodes, const float* __restrict__ agg,
    const float* __restrict__ g,
    const unsigned* __restrict__ Wp0, const unsigned* __restrict__ Wp1,
    const unsigned* __restrict__ Wp2, const unsigned* __restrict__ Wp3,
    const float* __restrict__ b0, const float* __restrict__ b1,
    const float* __restrict__ b2, const float* __restrict__ b3,
    float* __restrict__ out_nodes, float* __restrict__ node_sum, int N)
{
  __shared__ __align__(16) unsigned short buf0[64 * 162];
  __shared__ __align__(16) unsigned short buf1[64 * 130];
  __shared__ __align__(16) unsigned short buf2[64 * 130];
  __shared__ float colsum[64];
  const int tid = threadIdx.x;
  const int nb  = blockIdx.x * 64;
  if (tid < 64) colsum[tid] = 0.f;

  // gather: 160 cols = 40 quads (16 | 8 | 16)
  for (int idx = tid; idx < 64 * 40; idx += 256) {
    const int r = idx / 40, q = idx % 40;
    const int ni = nb + r;
    const int nc = ni < N ? ni : 0;
    float4 v;
    if (q < 16)      v = ((const float4*)(nodes + (size_t)nc * 64))[q];
    else if (q < 24) v = ((const float4*)(agg + (size_t)nc * 32))[q - 16];
    else             v = ((const float4*)g)[q - 24];
    pack4(&buf0[r * 162 + q * 4], v);
  }
  __syncthreads();

  const int w = tid >> 5, lane = tid & 31;
  layer64x128<160>(buf0, 162, buf1, 130, Wp0, b0, w, lane); __syncthreads();
  layer64x128<128>(buf1, 130, buf2, 130, Wp1, b1, w, lane); __syncthreads();
  layer64x128<128>(buf2, 130, buf1, 130, Wp2, b2, w, lane); __syncthreads();

  // final layer: 128 -> 64, fp32, no relu. Waves 0..3 active (cb<64).
  {
    const int half = lane >> 4, row = lane & 15;
    const int cb = w * 16;
    if (cb < 64) {
      v8f acc0 = {}, acc1 = {}, acc2 = {}, acc3 = {};
#pragma unroll
      for (int k0 = 0; k0 < 128; k0 += 32) {
        BFrag B;
#pragma unroll
        for (int v = 0; v < 8; ++v)
          B.u[v] = Wp3[((k0 + klA(v, half)) >> 1) * 64 + cb + row];
        BFrag A0, A1, A2, A3;
#pragma unroll
        for (int v = 0; v < 8; ++v) {
          const int ka = k0 + klA(v, half);
          A0.u[v] = *(const unsigned*)(buf1 + (0 * 16 + row) * 130 + ka);
          A1.u[v] = *(const unsigned*)(buf1 + (1 * 16 + row) * 130 + ka);
          A2.u[v] = *(const unsigned*)(buf1 + (2 * 16 + row) * 130 + ka);
          A3.u[v] = *(const unsigned*)(buf1 + (3 * 16 + row) * 130 + ka);
        }
        acc0 = __builtin_amdgcn_wmma_f32_16x16x32_bf16(false, A0.v, false, B.v, (short)0, acc0, false, false);
        acc1 = __builtin_amdgcn_wmma_f32_16x16x32_bf16(false, A1.v, false, B.v, (short)0, acc1, false, false);
        acc2 = __builtin_amdgcn_wmma_f32_16x16x32_bf16(false, A2.v, false, B.v, (short)0, acc2, false, false);
        acc3 = __builtin_amdgcn_wmma_f32_16x16x32_bf16(false, A3.v, false, B.v, (short)0, acc3, false, false);
      }
      const int n0 = cb + row;
      const float bb = b3[n0];
      float s = 0.f;
      v8f* accs[4] = { &acc0, &acc1, &acc2, &acc3 };
#pragma unroll
      for (int r = 0; r < 4; ++r) {
#pragma unroll
        for (int v = 0; v < 8; ++v) {
          const int m = r * 16 + half * 8 + v;
          const int ni = nb + m;
          if (ni < N) {
            const float x = (*accs[r])[v] + bb;
            out_nodes[(size_t)ni * 64 + n0] = x;
            s += x;
          }
        }
      }
      atomicAdd(&colsum[n0], s);
    }
  }
  __syncthreads();
  if (tid < 64) atomicAdd(&node_sum[tid], colsum[tid]);
}

// ---------------------------------------------------------------------------
// Global MLP: single row, scalar fp32 (trivial FLOPs)
// ---------------------------------------------------------------------------
__global__ __launch_bounds__(256) void global_mlp_kernel(
    const float* __restrict__ g,
    const float* __restrict__ node_sum, const float* __restrict__ edge_sum,
    const float* __restrict__ W0, const float* __restrict__ b0,
    const float* __restrict__ W1, const float* __restrict__ b1,
    const float* __restrict__ W2, const float* __restrict__ b2,
    const float* __restrict__ W3, const float* __restrict__ b3,
    float* __restrict__ outg, float invN, float invE)
{
  __shared__ float x[160];
  __shared__ float h1[128];
  __shared__ float h2[128];
  const int t = threadIdx.x;
  if (t < 64)       x[t] = g[t];
  else if (t < 128) x[t] = node_sum[t - 64] * invN;
  else if (t < 160) x[t] = edge_sum[t - 128] * invE;
  __syncthreads();
  if (t < 128) {
    float a = b0[t];
    for (int k = 0; k < 160; ++k) a += x[k] * W0[k * 128 + t];
    h1[t] = a > 0.f ? a : 0.f;
  }
  __syncthreads();
  if (t < 128) {
    float a = b1[t];
    for (int k = 0; k < 128; ++k) a += h1[k] * W1[k * 128 + t];
    h2[t] = a > 0.f ? a : 0.f;
  }
  __syncthreads();
  if (t < 128) {
    float a = b2[t];
    for (int k = 0; k < 128; ++k) a += h2[k] * W2[k * 128 + t];
    h1[t] = a > 0.f ? a : 0.f;
  }
  __syncthreads();
  if (t < 64) {
    float a = b3[t];
    for (int k = 0; k < 128; ++k) a += h1[k] * W3[k * 64 + t];
    outg[t] = a;
  }
}

// ---------------------------------------------------------------------------
// Weight packing: row-major fp32 [K][Nc] -> u32[(K/2)][Nc] of packed bf16 pairs
// ---------------------------------------------------------------------------
__global__ __launch_bounds__(256) void pack_w_kernel(
    const float* __restrict__ W, unsigned* __restrict__ dst, int K, int Nc)
{
  const int i = blockIdx.x * 256 + threadIdx.x;
  const int total = (K / 2) * Nc;
  if (i >= total) return;
  const int kp = i / Nc, n = i % Nc;
  const unsigned lo = f2bf(W[(size_t)(2 * kp) * Nc + n]);
  const unsigned hi = f2bf(W[(size_t)(2 * kp + 1) * Nc + n]);
  dst[i] = lo | (hi << 16);
}

__global__ __launch_bounds__(256) void zero_f_kernel(float* __restrict__ p, long n)
{
  long i = (long)blockIdx.x * blockDim.x + threadIdx.x;
  const long stride = (long)gridDim.x * blockDim.x;
  for (; i < n; i += stride) p[i] = 0.f;
}

// ---------------------------------------------------------------------------
extern "C" void kernel_launch(void* const* d_in, const int* in_sizes, int n_in,
                              void* d_out, int out_size, void* d_ws, size_t ws_size,
                              hipStream_t stream)
{
  const float* nodes = (const float*)d_in[0];
  const float* edges = (const float*)d_in[1];
  const float* g     = (const float*)d_in[2];
  const float* eW[4] = {(const float*)d_in[3],  (const float*)d_in[5],
                        (const float*)d_in[7],  (const float*)d_in[9]};
  const float* eb[4] = {(const float*)d_in[4],  (const float*)d_in[6],
                        (const float*)d_in[8],  (const float*)d_in[10]};
  const float* nW[4] = {(const float*)d_in[11], (const float*)d_in[13],
                        (const float*)d_in[15], (const float*)d_in[17]};
  const float* nb[4] = {(const float*)d_in[12], (const float*)d_in[14],
                        (const float*)d_in[16], (const float*)d_in[18]};
  const float* gW[4] = {(const float*)d_in[19], (const float*)d_in[21],
                        (const float*)d_in[23], (const float*)d_in[25]};
  const float* gb[4] = {(const float*)d_in[20], (const float*)d_in[22],
                        (const float*)d_in[24], (const float*)d_in[26]};
  const int* senders   = (const int*)d_in[27];
  const int* receivers = (const int*)d_in[28];

  const int N = in_sizes[0] / 64;
  const int E = in_sizes[1] / 32;

  // workspace layout (u32 units for packed weights, then fp32 scratch)
  unsigned* eWp0 = (unsigned*)d_ws;           // (224/2)*128 = 14336
  unsigned* eWp1 = eWp0 + 14336;              // 64*128 = 8192
  unsigned* eWp2 = eWp1 + 8192;               // 8192
  unsigned* eWp3 = eWp2 + 8192;               // 64*32 = 2048
  unsigned* nWp0 = eWp3 + 2048;               // (160/2)*128 = 10240
  unsigned* nWp1 = nWp0 + 10240;              // 8192
  unsigned* nWp2 = nWp1 + 8192;               // 8192
  unsigned* nWp3 = nWp2 + 8192;               // 64*64 = 4096
  float* agg      = (float*)(nWp3 + 4096);    // N*32 fp32
  float* edge_sum = agg + (size_t)N * 32;     // 32
  float* node_sum = edge_sum + 32;            // 64

  float* out_edges = (float*)d_out;
  float* out_nodes = out_edges + (size_t)E * 32;
  float* out_g     = out_nodes + (size_t)N * 64;

  // 1. pack weights to bf16 K-pair layout
  auto pack = [&](const float* W, unsigned* dst, int K, int Nc) {
    const int tot = (K / 2) * Nc;
    pack_w_kernel<<<(tot + 255) / 256, 256, 0, stream>>>(W, dst, K, Nc);
  };
  pack(eW[0], eWp0, 224, 128);
  pack(eW[1], eWp1, 128, 128);
  pack(eW[2], eWp2, 128, 128);
  pack(eW[3], eWp3, 128, 32);
  pack(nW[0], nWp0, 160, 128);
  pack(nW[1], nWp1, 128, 128);
  pack(nW[2], nWp2, 128, 128);
  pack(nW[3], nWp3, 128, 64);

  // 2. zero agg + mean accumulators
  zero_f_kernel<<<1024, 256, 0, stream>>>(agg, (long)N * 32 + 96);

  // 3. edge MLP (+ segment-sum scatter, edge column sums)
  edge_kernel<<<(E + 63) / 64, 256, 0, stream>>>(
      nodes, edges, g, senders, receivers,
      eWp0, eWp1, eWp2, eWp3, eb[0], eb[1], eb[2], eb[3],
      out_edges, agg, edge_sum, E);

  // 4. node MLP (+ node column sums)
  node_kernel<<<(N + 63) / 64, 256, 0, stream>>>(
      nodes, agg, g,
      nWp0, nWp1, nWp2, nWp3, nb[0], nb[1], nb[2], nb[3],
      out_nodes, node_sum, N);

  // 5. global MLP
  global_mlp_kernel<<<1, 256, 0, stream>>>(
      g, node_sum, edge_sum,
      gW[0], gb[0], gW[1], gb[1], gW[2], gb[2], gW[3], gb[3],
      out_g, 1.0f / (float)N, 1.0f / (float)E);
}
